// HS_MSA_292057776446
// MI455X (gfx1250) — compile-verified, matrix operand order checked by reading, not run on hardware
//
#include <hip/hip_runtime.h>
#include <hip/hip_bf16.h>

typedef __attribute__((ext_vector_type(16))) __bf16 bf16x16;
typedef __attribute__((ext_vector_type(8)))  __bf16 bf16x8;
typedef __attribute__((ext_vector_type(4)))  __bf16 bf16x4;
typedef __attribute__((ext_vector_type(8)))  float  f32x8;
typedef __attribute__((ext_vector_type(4)))  unsigned int u32x4;
typedef __attribute__((ext_vector_type(8)))  int        i32x8;
typedef __attribute__((ext_vector_type(4)))  int        i32x4;

#define WMMA_BF16(A_, B_, C_) \
  __builtin_amdgcn_wmma_f32_16x16x32_bf16(false, (A_), false, (B_), (short)0, (C_), false, false)

#if defined(__has_builtin)
#if __has_builtin(__builtin_amdgcn_tensor_load_to_lds) && \
    __has_builtin(__builtin_amdgcn_s_wait_tensorcnt)
#define HAVE_TDM 1
#endif
#endif

// ---------------------------------------------------------------------------
// Kernel 1: qkv = x @ [Wq | Wkv]   (M=131072, K=224, N=672), bf16 out
// block: 256 thr (8 waves). Tile: 128 rows x 48 cols. K-tile = 32.
// ---------------------------------------------------------------------------
__global__ __launch_bounds__(256)
void qkv_gemm(const float* __restrict__ x, const float* __restrict__ Wq,
              const float* __restrict__ Wkv, __bf16* __restrict__ qkv) {
  __shared__ __bf16 As[128 * 32];   // A tile, row-major in k
  __shared__ __bf16 Bs[48 * 32];    // B tile transposed: [n][k]
  const int bn = blockIdx.x, bm = blockIdx.y;
  const int tid  = threadIdx.x;
  const int wave = tid >> 5, lane = tid & 31;
  const int hl = lane >> 4, l15 = lane & 15;
  const long rowBase = (long)bm * 128;

  f32x8 acc[3];
  #pragma unroll
  for (int t = 0; t < 3; ++t) acc[t] = (f32x8){};

  for (int kt = 0; kt < 7; ++kt) {
    const int k0 = kt * 32;
    // stage A: 128x32 fp32 -> bf16, 16B global loads, 8B LDS stores
    #pragma unroll
    for (int it = 0; it < 4; ++it) {
      int i = tid + it * 256;            // 1024 float4 total
      int m = i >> 3, c = i & 7;
      const float4 v = ((const float4*)(x + (rowBase + m) * 224 + k0))[c];
      bf16x4 w = { (__bf16)v.x, (__bf16)v.y, (__bf16)v.z, (__bf16)v.w };
      *(bf16x4*)(As + m * 32 + c * 4) = w;
    }
    if (kt < 6) __builtin_prefetch(&x[(rowBase + (tid >> 1)) * 224 + k0 + 32], 0, 1);
    // stage B^T: 48x32, vectorized along n (4 cols per 16B load)
    for (int i = tid; i < 12 * 32; i += 256) {
      int nn4 = i % 12, kk = i / 12;
      int ng  = bn * 48 + nn4 * 4;
      const float4 v = (ng < 224)
          ? *(const float4*)(Wq  + (k0 + kk) * 224 + ng)
          : *(const float4*)(Wkv + (k0 + kk) * 448 + (ng - 224));
      Bs[(nn4 * 4 + 0) * 32 + kk] = (__bf16)v.x;
      Bs[(nn4 * 4 + 1) * 32 + kk] = (__bf16)v.y;
      Bs[(nn4 * 4 + 2) * 32 + kk] = (__bf16)v.z;
      Bs[(nn4 * 4 + 3) * 32 + kk] = (__bf16)v.w;
    }
    __syncthreads();

    bf16x16 a;
    const int m  = wave * 16 + l15;
    const int kb = hl * 8;
    #pragma unroll
    for (int j = 0; j < 8; ++j) {
      a[j]     = As[m * 32 + kb + j];
      a[8 + j] = As[m * 32 + 16 + kb + j];
    }
    #pragma unroll
    for (int t = 0; t < 3; ++t) {
      bf16x16 bv;
      int n = t * 16 + l15;
      #pragma unroll
      for (int j = 0; j < 8; ++j) {
        bv[j]     = Bs[n * 32 + kb + j];
        bv[8 + j] = Bs[n * 32 + 16 + kb + j];
      }
      acc[t] = WMMA_BF16(a, bv, acc[t]);
    }
    __syncthreads();
  }
  #pragma unroll
  for (int t = 0; t < 3; ++t) {
    long col = (long)bn * 48 + t * 16 + l15;
    #pragma unroll
    for (int r = 0; r < 8; ++r) {
      long row = rowBase + wave * 16 + r + 8 * hl;
      qkv[row * 672 + col] = (__bf16)acc[t][r];
    }
  }
}

// ---------------------------------------------------------------------------
// Kernel 2: local window attention. block = (b*16 + window), 4 waves = 4 heads
// ---------------------------------------------------------------------------
__global__ __launch_bounds__(128)
void local_attn(const __bf16* __restrict__ qkv, const float* __restrict__ pos1,
                __bf16* __restrict__ aout) {
  __shared__ __bf16 lds[4 * 6144];
  const int tid  = threadIdx.x;
  const int h    = tid >> 5, lane = tid & 31;
  const int hl   = lane >> 4, l15 = lane & 15;
  const int bw   = blockIdx.x;
  const int b    = bw >> 4, win = bw & 15;
  const int wy   = win >> 2, wx = win & 3;
  __bf16* Q  = lds + h * 6144;
  __bf16* K  = Q + 2048;
  __bf16* Vt = K + 2048;

  // Q,K: 64 tokens x 8 chunks of 4 dims (chunk 7 = zero pad 28..31)
  #pragma unroll
  for (int it = 0; it < 16; ++it) {
    int i = lane + it * 32;              // 512
    int tok = i >> 3, c = i & 7;
    int iy = tok >> 3, ix = tok & 7;
    long row = (long)b * 1024 + (wy * 8 + iy) * 32 + (wx * 8 + ix);
    bf16x4 qv = {}, kv = {};
    if (c < 7) {
      qv = *(const bf16x4*)(qkv + row * 672 + h * 28 + c * 4);
      kv = *(const bf16x4*)(qkv + row * 672 + 224 + h * 28 + c * 4);
    }
    *(bf16x4*)(Q + tok * 32 + c * 4) = qv;
    *(bf16x4*)(K + tok * 32 + c * 4) = kv;
  }
  // Vt: [dim][token] transpose, scalar
  for (int i = lane; i < 2048; i += 32) {
    int d = i >> 6, tok = i & 63;
    int iy = tok >> 3, ix = tok & 7;
    long row = (long)b * 1024 + (wy * 8 + iy) * 32 + (wx * 8 + ix);
    Vt[d * 64 + tok] = (d < 28) ? qkv[row * 672 + 448 + h * 28 + d]
                                : (__bf16)0.f;
  }
  __syncthreads();

  // sim = Q @ K^T : 4x4 tiles of 16x16, K=32
  f32x8 s[4][4];
  #pragma unroll
  for (int tr = 0; tr < 4; ++tr)
    #pragma unroll
    for (int tc = 0; tc < 4; ++tc) s[tr][tc] = (f32x8){};
  const int kb = hl * 8;
  #pragma unroll
  for (int tr = 0; tr < 4; ++tr) {
    bf16x16 a;
    int m = tr * 16 + l15;
    #pragma unroll
    for (int j = 0; j < 8; ++j) { a[j] = Q[m * 32 + kb + j]; a[8 + j] = Q[m * 32 + 16 + kb + j]; }
    #pragma unroll
    for (int tc = 0; tc < 4; ++tc) {
      bf16x16 bv;
      int n = tc * 16 + l15;
      #pragma unroll
      for (int j = 0; j < 8; ++j) { bv[j] = K[n * 32 + kb + j]; bv[8 + j] = K[n * 32 + 16 + kb + j]; }
      s[tr][tc] = WMMA_BF16(a, bv, s[tr][tc]);
    }
  }

  const float scale = 0.1889822365046136f;  // 28^-0.5
  #pragma unroll
  for (int tr = 0; tr < 4; ++tr) {
    #pragma unroll
    for (int r = 0; r < 8; ++r) {
      int m = tr * 16 + r + 8 * hl;
      float lv[4];
      float mx = -1e30f;
      #pragma unroll
      for (int tc = 0; tc < 4; ++tc) {
        int n = tc * 16 + l15;
        float v = s[tr][tc][r] * scale + pos1[(h * 64 + m) * 64 + n];
        lv[tc] = v; mx = fmaxf(mx, v);
      }
      for (int off = 8; off >= 1; off >>= 1) mx = fmaxf(mx, __shfl_xor(mx, off, 32));
      float sum = 0.f;
      #pragma unroll
      for (int tc = 0; tc < 4; ++tc) { lv[tc] = __expf(lv[tc] - mx); sum += lv[tc]; }
      for (int off = 8; off >= 1; off >>= 1) sum += __shfl_xor(sum, off, 32);
      float inv = 1.f / sum;
      #pragma unroll
      for (int tc = 0; tc < 4; ++tc) s[tr][tc][r] = lv[tc] * inv;
    }
  }
  __syncthreads();

  __bf16* P = Q;  // 64x64 bf16 over Q+K region
  #pragma unroll
  for (int tr = 0; tr < 4; ++tr)
    #pragma unroll
    for (int r = 0; r < 8; ++r) {
      int m = tr * 16 + r + 8 * hl;
      #pragma unroll
      for (int tc = 0; tc < 4; ++tc)
        P[m * 64 + tc * 16 + l15] = (__bf16)s[tr][tc][r];
    }
  __syncthreads();

  f32x8 o[4][2];
  #pragma unroll
  for (int tr = 0; tr < 4; ++tr)
    #pragma unroll
    for (int tc = 0; tc < 2; ++tc) o[tr][tc] = (f32x8){};
  #pragma unroll
  for (int ks = 0; ks < 2; ++ks) {
    int kk = ks * 32 + kb;
    #pragma unroll
    for (int tr = 0; tr < 4; ++tr) {
      bf16x16 a;
      int m = tr * 16 + l15;
      #pragma unroll
      for (int j = 0; j < 8; ++j) { a[j] = P[m * 64 + kk + j]; a[8 + j] = P[m * 64 + kk + 16 + j]; }
      #pragma unroll
      for (int tc = 0; tc < 2; ++tc) {
        bf16x16 bv;
        int n = tc * 16 + l15;
        #pragma unroll
        for (int j = 0; j < 8; ++j) { bv[j] = Vt[n * 64 + kk + j]; bv[8 + j] = Vt[n * 64 + kk + 16 + j]; }
        o[tr][tc] = WMMA_BF16(a, bv, o[tr][tc]);
      }
    }
  }
  #pragma unroll
  for (int tr = 0; tr < 4; ++tr)
    #pragma unroll
    for (int tc = 0; tc < 2; ++tc)
      #pragma unroll
      for (int r = 0; r < 8; ++r) {
        int m = tr * 16 + r + 8 * hl;
        int d = tc * 16 + l15;
        if (d < 28)
          aout[((long)bw * 64 + m) * 224 + h * 28 + d] = (__bf16)o[tr][tc][r];
      }
}

// ---------------------------------------------------------------------------
// Kernel 3: global (cross-window) attention. block = (p, b), 4 waves = heads
// ---------------------------------------------------------------------------
__global__ __launch_bounds__(128)
void global_attn(const __bf16* __restrict__ qkv, const float* __restrict__ pos2,
                 __bf16* __restrict__ aout) {
  __shared__ __bf16 lds[4 * 2048];
  const int tid  = threadIdx.x;
  const int h    = tid >> 5, lane = tid & 31;
  const int hl   = lane >> 4, l15 = lane & 15;
  const int p    = blockIdx.x;
  const int b    = blockIdx.y;
  const int iy = p >> 3, ix = p & 7;
  __bf16* Q  = lds + h * 2048;
  __bf16* K  = Q + 512;
  __bf16* Vt = K + 512;

  #pragma unroll
  for (int it = 0; it < 4; ++it) {
    int i = lane + it * 32;              // 128 = 16 windows x 8 chunks
    int w = i >> 3, c = i & 7;
    int wy = w >> 2, wx = w & 3;
    long row = (long)b * 1024 + (wy * 8 + iy) * 32 + (wx * 8 + ix);
    bf16x4 qv = {}, kv = {};
    if (c < 7) {
      qv = *(const bf16x4*)(qkv + row * 672 + 112 + h * 28 + c * 4);
      kv = *(const bf16x4*)(qkv + row * 672 + 336 + h * 28 + c * 4);
    }
    *(bf16x4*)(Q + w * 32 + c * 4) = qv;
    *(bf16x4*)(K + w * 32 + c * 4) = kv;
  }
  for (int i = lane; i < 1024; i += 32) {   // Vt[d][w], both dims padded
    int d = i >> 5, w = i & 31;
    __bf16 v = (__bf16)0.f;
    if (d < 28 && w < 16) {
      int wy = w >> 2, wx = w & 3;
      long row = (long)b * 1024 + (wy * 8 + iy) * 32 + (wx * 8 + ix);
      v = qkv[row * 672 + 560 + h * 28 + d];
    }
    Vt[d * 32 + w] = v;
  }
  __syncthreads();

  const int kb = hl * 8;
  f32x8 s = (f32x8){};
  {
    bf16x16 a, bv;
    #pragma unroll
    for (int j = 0; j < 8; ++j) {
      a[j]      = Q[l15 * 32 + kb + j];
      a[8 + j]  = Q[l15 * 32 + 16 + kb + j];
      bv[j]     = K[l15 * 32 + kb + j];
      bv[8 + j] = K[l15 * 32 + 16 + kb + j];
    }
    s = WMMA_BF16(a, bv, s);
  }

  const float scale = 0.1889822365046136f;
  #pragma unroll
  for (int r = 0; r < 8; ++r) {
    int m = r + 8 * hl;
    float v = s[r] * scale + pos2[(h * 16 + m) * 16 + l15];
    float mx = v;
    for (int off = 8; off >= 1; off >>= 1) mx = fmaxf(mx, __shfl_xor(mx, off, 32));
    float e = __expf(v - mx);
    float sum = e;
    for (int off = 8; off >= 1; off >>= 1) sum += __shfl_xor(sum, off, 32);
    s[r] = e / sum;
  }
  __syncthreads();

  __bf16* P = Q;
  #pragma unroll
  for (int r = 0; r < 8; ++r) {
    int m = r + 8 * hl;
    P[m * 32 + l15]      = (__bf16)s[r];
    P[m * 32 + 16 + l15] = (__bf16)0.f;
  }
  __syncthreads();

  f32x8 o[2];
  o[0] = (f32x8){}; o[1] = (f32x8){};
  {
    bf16x16 a;
    #pragma unroll
    for (int j = 0; j < 8; ++j) {
      a[j]     = P[l15 * 32 + kb + j];
      a[8 + j] = P[l15 * 32 + 16 + kb + j];
    }
    #pragma unroll
    for (int tc = 0; tc < 2; ++tc) {
      bf16x16 bv;
      int n = tc * 16 + l15;
      #pragma unroll
      for (int j = 0; j < 8; ++j) {
        bv[j]     = Vt[n * 32 + kb + j];
        bv[8 + j] = Vt[n * 32 + 16 + kb + j];
      }
      o[tc] = WMMA_BF16(a, bv, o[tc]);
    }
  }
  #pragma unroll
  for (int tc = 0; tc < 2; ++tc)
    #pragma unroll
    for (int r = 0; r < 8; ++r) {
      int m = r + 8 * hl;
      int d = tc * 16 + l15;
      if (d < 28)
        aout[(((long)b * 16 + m) * 64 + p) * 224 + 112 + h * 28 + d] =
            (__bf16)o[tc][r];
    }
}

// ---------------------------------------------------------------------------
// Kernel 4: out = attn_out @ Wout + bout, window-reverse on store (fp32 out)
// A tile (128x32 bf16, stride 224) staged via Tensor Data Mover when available
// ---------------------------------------------------------------------------
__global__ __launch_bounds__(256)
void out_gemm(const __bf16* __restrict__ ain, const float* __restrict__ Wout,
              const float* __restrict__ bout, float* __restrict__ out) {
  __shared__ __bf16 As[128 * 32];
  __shared__ __bf16 Bs[224 * 32];
  const int bm = blockIdx.x;
  const int tid  = threadIdx.x;
  const int wave = tid >> 5, lane = tid & 31;
  const int hl = lane >> 4, l15 = lane & 15;
  const long rowBase = (long)bm * 128;

  f32x8 acc[14];
  #pragma unroll
  for (int t = 0; t < 14; ++t) acc[t] = (f32x8){};

  for (int kt = 0; kt < 7; ++kt) {
    const int k0 = kt * 32;
#if defined(HAVE_TDM)
    if (wave == 0) {
      // TDM descriptor: 2-D bf16 tile 32x128, row stride 224 elements
      unsigned long long ga =
          (unsigned long long)(uintptr_t)(ain + rowBase * 224 + k0);
      unsigned int la = (unsigned int)(uintptr_t)(void*)As;
      u32x4 g0 = { 1u,                       // count=1, user mode
                   la,                       // lds_addr
                   (unsigned int)ga,         // global_addr[31:0]
                   (unsigned int)((ga >> 32) & 0x1FFFFFFu) | 0x80000000u }; // addr[56:32] | type=2
      i32x8 g1 = { 0x00010000,               // data_size=1 (2B)
                   224 << 16,                // tensor_dim0 = 224
                   128 << 16,                // tensor_dim1 = 128
                   2 | (32 << 16),           // tensor_dim1 hi | tile_dim0=32
                   128,                      // tile_dim1=128, tile_dim2=0
                   224,                      // tensor_dim0_stride = 224
                   0, 0 };
      i32x4 gz = { 0, 0, 0, 0 };
#if __clang_major__ >= 23
      i32x8 g4 = { 0, 0, 0, 0, 0, 0, 0, 0 };
      __builtin_amdgcn_tensor_load_to_lds(g0, g1, gz, gz, g4, 0);
#else
      __builtin_amdgcn_tensor_load_to_lds(g0, g1, gz, gz, 0);
#endif
    }
#else
    // fallback: vectorized manual staging (bf16x8 = 16B both sides)
    #pragma unroll
    for (int it = 0; it < 2; ++it) {
      int i = tid + it * 256;              // 512 x 16B
      int m = i >> 2, c = i & 3;
      bf16x8 v = *(const bf16x8*)(ain + (rowBase + m) * 224 + k0 + c * 8);
      *(bf16x8*)(As + m * 32 + c * 8) = v;
    }
#endif
    if (kt < 6) __builtin_prefetch(&ain[(rowBase + (tid >> 1)) * 224 + k0 + 32], 0, 1);
    // stage B^T: 224x32, 16B loads of 4 consecutive cols
    #pragma unroll
    for (int it = 0; it < 7; ++it) {
      int i = tid + it * 256;              // 1792 = 56 col-groups x 32 k
      int nn4 = i % 56, kk = i / 56;
      const float4 v = *(const float4*)(Wout + (k0 + kk) * 224 + nn4 * 4);
      Bs[(nn4 * 4 + 0) * 32 + kk] = (__bf16)v.x;
      Bs[(nn4 * 4 + 1) * 32 + kk] = (__bf16)v.y;
      Bs[(nn4 * 4 + 2) * 32 + kk] = (__bf16)v.z;
      Bs[(nn4 * 4 + 3) * 32 + kk] = (__bf16)v.w;
    }
#if defined(HAVE_TDM)
    if (wave == 0) __builtin_amdgcn_s_wait_tensorcnt(0);
#endif
    __syncthreads();

    bf16x16 a;
    const int m  = wave * 16 + l15;
    const int kb = hl * 8;
    #pragma unroll
    for (int j = 0; j < 8; ++j) {
      a[j]     = As[m * 32 + kb + j];
      a[8 + j] = As[m * 32 + 16 + kb + j];
    }
    #pragma unroll
    for (int t = 0; t < 14; ++t) {
      bf16x16 bv;
      int n = t * 16 + l15;
      #pragma unroll
      for (int j = 0; j < 8; ++j) {
        bv[j]     = Bs[n * 32 + kb + j];
        bv[8 + j] = Bs[n * 32 + 16 + kb + j];
      }
      acc[t] = WMMA_BF16(a, bv, acc[t]);
    }
    __syncthreads();
  }

  #pragma unroll
  for (int t = 0; t < 14; ++t) {
    int n = t * 16 + l15;
    float bias = bout[n];
    #pragma unroll
    for (int r = 0; r < 8; ++r) {
      long Rm = rowBase + wave * 16 + r + 8 * hl;
      int b   = (int)(Rm >> 10);
      int rem = (int)(Rm & 1023);
      int win = rem >> 6, tok = rem & 63;
      int wy = win >> 2, wx = win & 3;
      int iy = tok >> 3, ix = tok & 7;
      int y = wy * 8 + iy, xx = wx * 8 + ix;
      out[(((long)b * 32 + y) * 32 + xx) * 224 + n] = acc[t][r] + bias;
    }
  }
}

// ---------------------------------------------------------------------------
extern "C" void kernel_launch(void* const* d_in, const int* in_sizes, int n_in,
                              void* d_out, int out_size, void* d_ws, size_t ws_size,
                              hipStream_t stream) {
  const float* x    = (const float*)d_in[0];
  const float* Wq   = (const float*)d_in[1];
  const float* Wkv  = (const float*)d_in[2];
  const float* Wout = (const float*)d_in[3];
  const float* bout = (const float*)d_in[4];
  const float* pos1 = (const float*)d_in[5];
  const float* pos2 = (const float*)d_in[6];

  __bf16* qkv  = (__bf16*)d_ws;                       // 131072 x 672 bf16
  __bf16* aout = qkv + (size_t)131072 * 672;          // 131072 x 224 bf16
  float*  out  = (float*)d_out;

  qkv_gemm   <<<dim3(14, 1024), 256, 0, stream>>>(x, Wq, Wkv, qkv);
  local_attn <<<dim3(2048),     128, 0, stream>>>(qkv, pos1, aout);
  global_attn<<<dim3(64, 128),  128, 0, stream>>>(qkv, pos2, aout);
  out_gemm   <<<dim3(1024),     256, 0, stream>>>(aout, Wout, bout, out);
}
